// Decoder_66486093742303
// MI455X (gfx1250) — compile-verified, matrix operand order checked by reading, not run on hardware
//
#include <hip/hip_runtime.h>
#include <math.h>

// ---------------- types ----------------
typedef __attribute__((ext_vector_type(16))) __bf16 v16bf;
typedef __attribute__((ext_vector_type(8)))  float  v8f;

#define T64 64
#define H512 512
#define S512 512
#define E256 256
#define L50K 50000
#define NWG 16

// K-index map for 16-bit WMMA A/B fragments (ISA 7.12.2, 16-bit A 16x32):
// element i of the per-lane v16bf, lane-high bit hi = lane>>4.
__device__ __forceinline__ int kmap(int i, int hi) {
    return (i & 7) + 8 * (((i >> 3) << 1) + hi);
}

// ---------------- init ----------------
__global__ void init_k(int* done, float* losses, float* out) {
    int tid = threadIdx.x;
    if (tid < T64 + 1) done[tid] = 0;
    if (tid < T64)     losses[tid] = 0.f;
    if (tid == 0)      out[0] = 0.f;
}

// ---------------- fused attention: emb -> q -> softmax(enc@q) -> ctx ----------------
// grid = 64 (one block per timestep), block = 512 threads
__global__ void attn_ctx(const float* __restrict__ enc, const float* __restrict__ emb,
                         const float* __restrict__ Wattn, const float* __restrict__ battn,
                         const int* __restrict__ words, __bf16* __restrict__ ctxb) {
    __shared__ float e[E256];
    __shared__ float q[H512];
    __shared__ float sc[S512];
    __shared__ float red[16];
    __shared__ float bmax, bsum;
    const int t = blockIdx.x, tid = threadIdx.x;
    const int w = words[t];
    if (tid < E256) e[tid] = emb[(size_t)w * E256 + tid];
    __syncthreads();
    // q = W_attn @ e + b_attn   (thread tid = output row)
    {
        float a = battn[tid];
        const float* wr = Wattn + (size_t)tid * E256;
        #pragma unroll 8
        for (int c = 0; c < E256; ++c) a += wr[c] * e[c];
        q[tid] = a;
    }
    __syncthreads();
    // scores[s] = enc[s,:] . q
    float s = 0.f;
    {
        const float* er = enc + (size_t)tid * H512;
        #pragma unroll 8
        for (int h = 0; h < H512; ++h) s += er[h] * q[h];
    }
    // block max (deterministic)
    float m = s;
    for (int off = 16; off; off >>= 1) m = fmaxf(m, __shfl_xor(m, off));
    if ((tid & 31) == 0) red[tid >> 5] = m;
    __syncthreads();
    if (tid == 0) {
        float mm = red[0];
        for (int i = 1; i < 16; ++i) mm = fmaxf(mm, red[i]);
        bmax = mm;
    }
    __syncthreads();
    float p = expf(s - bmax);
    sc[tid] = p;
    float ssum = p;
    for (int off = 16; off; off >>= 1) ssum += __shfl_xor(ssum, off);
    if ((tid & 31) == 0) red[tid >> 5] = ssum;
    __syncthreads();
    if (tid == 0) {
        float tot = 0.f;
        for (int i = 0; i < 16; ++i) tot += red[i];
        bsum = tot;
    }
    __syncthreads();
    const float inv = 1.f / bsum;
    // ctx[h] = sum_s alpha[s] * enc[s,h]   (thread tid = h, coalesced)
    float cacc = 0.f;
    #pragma unroll 8
    for (int s2 = 0; s2 < S512; ++s2) cacc += sc[s2] * enc[(size_t)s2 * H512 + tid];
    ctxb[(size_t)t * H512 + tid] = (__bf16)(cacc * inv);
}

// ---------------- WMMA GEMM: C[64,N] = A_bf16[64,K] * B_f32[N,K]^T + bias ----------------
// block = 256 threads (8 waves); each wave owns one 16-wide N strip; block covers 128 cols.
// A is staged into LDS pre-swizzled into WMMA fragment order (32B/lane contiguous).
__global__ void gemm_bf16_wmma(const __bf16* __restrict__ A, const float* __restrict__ B,
                               const float* __restrict__ bias, float* __restrict__ C,
                               int N, int K) {
    extern __shared__ __bf16 alds[];       // [4][K/32][32 lanes][16 elems]
    const int tid = threadIdx.x;
    const int lane = tid & 31;
    const int wave = tid >> 5;
    const int hi = lane >> 4;
    const int lo = lane & 15;
    const int KT = K >> 5;                 // K-steps of 32
    // stage A fragments
    const int total = 4 * KT * 32 * 16;
    for (int idx = tid; idx < total; idx += 256) {
        int i  = idx & 15;
        int l  = (idx >> 4) & 31;
        int f  = idx >> 9;
        int kt = f % KT;
        int mt = f / KT;
        int m  = mt * 16 + (l & 15);
        int k  = kt * 32 + kmap(i, l >> 4);
        alds[idx] = A[(size_t)m * K + k];
    }
    __syncthreads();

    const int n0   = (blockIdx.x * 8 + wave) * 16;
    const int ncol = n0 + lo;
    const int nc   = ncol < N ? ncol : (N - 1);      // clamp OOB loads
    const float* Brow = B + (size_t)nc * K;

    v8f acc[4];
    #pragma unroll
    for (int mt = 0; mt < 4; ++mt) acc[mt] = (v8f){0.f, 0.f, 0.f, 0.f, 0.f, 0.f, 0.f, 0.f};

    for (int kt = 0; kt < KT; ++kt) {
        v16bf bfrag;
        const int kb = kt * 32 + 8 * hi;
        #pragma unroll
        for (int i = 0; i < 8; ++i) bfrag[i]     = (__bf16)Brow[kb + i];
        #pragma unroll
        for (int i = 0; i < 8; ++i) bfrag[8 + i] = (__bf16)Brow[kb + 16 + i];
        #pragma unroll
        for (int mt = 0; mt < 4; ++mt) {
            const v16bf afrag =
                *(const v16bf*)(void*)(alds + ((size_t)(mt * KT + kt) * 32 + lane) * 16);
            acc[mt] = __builtin_amdgcn_wmma_f32_16x16x32_bf16(
                false, afrag, false, bfrag, (short)0, acc[mt], false, false);
        }
    }
    if (ncol < N) {
        const float bv = bias[ncol];
        #pragma unroll
        for (int mt = 0; mt < 4; ++mt) {
            #pragma unroll
            for (int r = 0; r < 8; ++r) {
                int m = mt * 16 + r + 8 * hi;
                C[(size_t)m * N + ncol] = acc[mt][r] + bv;
            }
        }
    }
}

// ---------------- sequential LSTM, W_hh sharded across 16 WGs (LDS-resident) ----------
// grid = 16, block = 256, dynamic LDS = 128*512*4 + 512*4 + 256*4 + 128*4 + 128*4 bytes
__global__ void lstm_seq(const float* __restrict__ Whh, const float* __restrict__ bhh,
                         const float* __restrict__ h0, const float* __restrict__ c0,
                         const float* __restrict__ gin, float* __restrict__ h_all,
                         __bf16* __restrict__ h2b, int* __restrict__ done) {
    extern __shared__ float sm[];
    float* wl   = sm;                    // [128][512] this WG's W_hh rows
    float* hb   = wl + 128 * 512;        // [512] current h
    float* part = hb + 512;              // [256] dot partials
    float* gate = part + 256;            // [128] gate pre-activations
    float* bl   = gate + 128;            // [128] b_hh slice
    const int tid = threadIdx.x;
    const int wg  = blockIdx.x;
    const int j0  = wg * 32;             // hidden units [j0, j0+32)
    // local row lr in [0,128): gate g = lr>>5, hidden j = j0 + (lr&31)
    for (int idx = tid; idx < 128 * 512; idx += 256) {
        int lr = idx >> 9, c = idx & 511;
        int grow = ((lr >> 5) * H512) + j0 + (lr & 31);
        wl[idx] = Whh[(size_t)grow * H512 + c];
    }
    if (tid < 128) {
        int grow = ((tid >> 5) * H512) + j0 + (tid & 31);
        bl[tid] = bhh[grow];
    }
    float cst = (tid < 32) ? c0[j0 + tid] : 0.f;
    __syncthreads();
    const int lr = tid >> 1, half = tid & 1;
    for (int t = 0; t < T64; ++t) {
        if (t == 0) {
            hb[tid] = h0[tid];
            hb[256 + tid] = h0[256 + tid];
        } else {
            if (tid == 0) {
                while (__hip_atomic_load(done + t, __ATOMIC_ACQUIRE,
                                         __HIP_MEMORY_SCOPE_AGENT) < NWG)
                    __builtin_amdgcn_s_sleep(1);
            }
            __syncthreads();
            hb[tid]       = h_all[(size_t)t * H512 + tid];
            hb[256 + tid] = h_all[(size_t)t * H512 + 256 + tid];
        }
        __syncthreads();
        // gates: 2 threads per row, 256 MACs each out of LDS
        {
            const float* wrow = wl + lr * 512 + half * 256;
            const float* hp   = hb + half * 256;
            float acc = 0.f;
            #pragma unroll 8
            for (int c = 0; c < 256; ++c) acc += wrow[c] * hp[c];
            part[tid] = acc;
        }
        __syncthreads();
        if (half == 0) {
            int grow = ((lr >> 5) * H512) + j0 + (lr & 31);
            gate[lr] = part[2 * lr] + part[2 * lr + 1] + gin[(size_t)t * 2048 + grow] + bl[lr];
        }
        __syncthreads();
        if (tid < 32) {
            float gi = gate[tid], gf = gate[32 + tid], gg = gate[64 + tid], go = gate[96 + tid];
            gi = 1.f / (1.f + expf(-gi));
            gf = 1.f / (1.f + expf(-gf));
            gg = tanhf(gg);
            go = 1.f / (1.f + expf(-go));
            cst = gf * cst + gi * gg;
            float h2 = go * tanhf(cst);
            h_all[(size_t)(t + 1) * H512 + j0 + tid] = h2;
            h2b[(size_t)t * H512 + j0 + tid] = (__bf16)h2;
        }
        __threadfence();
        __syncthreads();
        if (tid == 0)
            __hip_atomic_fetch_add(done + (t + 1), 1, __ATOMIC_RELEASE,
                                   __HIP_MEMORY_SCOPE_AGENT);
    }
}

// ---------------- per-step NLL via online log-softmax (deterministic) ----------------
__global__ void loss_k(const float* __restrict__ logits, const int* __restrict__ labels,
                       float* __restrict__ losses) {
    const int t = blockIdx.x, tid = threadIdx.x;
    const float* row = logits + (size_t)t * L50K;
    float m = -INFINITY, s = 0.f;
    for (int l = tid; l < L50K; l += 256) {
        float x = row[l];
        float nm = fmaxf(m, x);
        s = s * expf(m - nm) + expf(x - nm);
        m = nm;
    }
    for (int off = 16; off; off >>= 1) {
        float m2 = __shfl_xor(m, off), s2 = __shfl_xor(s, off);
        float nm = fmaxf(m, m2);
        s = s * expf(m - nm) + s2 * expf(m2 - nm);
        m = nm;
    }
    __shared__ float ms[8], ss[8];
    if ((tid & 31) == 0) { ms[tid >> 5] = m; ss[tid >> 5] = s; }
    __syncthreads();
    if (tid == 0) {
        float mm = ms[0], sm = ss[0];
        for (int i = 1; i < 8; ++i) {
            float nm = fmaxf(mm, ms[i]);
            sm = sm * expf(mm - nm) + ss[i] * expf(ms[i] - nm);
            mm = nm;
        }
        int lab = labels[t];
        losses[t] = (lab == 0) ? 0.f : -(row[lab] - mm - logf(sm));
    }
}

__global__ void sum_k(const float* __restrict__ losses, float* __restrict__ out) {
    __shared__ float buf[T64];
    buf[threadIdx.x] = losses[threadIdx.x];
    __syncthreads();
    for (int off = 32; off; off >>= 1) {
        if (threadIdx.x < off) buf[threadIdx.x] += buf[threadIdx.x + off];
        __syncthreads();
    }
    if (threadIdx.x == 0) out[0] = buf[0];
}

// ---------------- launch ----------------
extern "C" void kernel_launch(void* const* d_in, const int* in_sizes, int n_in,
                              void* d_out, int out_size, void* d_ws, size_t ws_size,
                              hipStream_t stream) {
    (void)in_sizes; (void)n_in; (void)out_size; (void)ws_size;
    const float* enc   = (const float*)d_in[0];
    const float* h0    = (const float*)d_in[1];
    const float* c0    = (const float*)d_in[2];
    const float* emb   = (const float*)d_in[3];
    const float* Wattn = (const float*)d_in[4];
    const float* battn = (const float*)d_in[5];
    const float* Wih   = (const float*)d_in[6];
    const float* Whh   = (const float*)d_in[7];
    const float* bih   = (const float*)d_in[8];
    const float* bhh   = (const float*)d_in[9];
    const float* Wout  = (const float*)d_in[10];
    const float* bout  = (const float*)d_in[11];
    const int*   words = (const int*)d_in[12];
    const int*   labels= (const int*)d_in[13];

    char* p = (char*)d_ws;
    auto alloc = [&](size_t bytes) -> void* {
        void* r = (void*)p;
        p += (bytes + 255) & ~(size_t)255;
        return r;
    };
    __bf16* ctxb   = (__bf16*)alloc((size_t)T64 * H512 * 2);
    float*  g_in   = (float*) alloc((size_t)T64 * 2048 * 4);
    float*  h_all  = (float*) alloc((size_t)(T64 + 1) * H512 * 4);
    __bf16* h2b    = (__bf16*)alloc((size_t)T64 * H512 * 2);
    float*  logits = (float*) alloc((size_t)T64 * L50K * 4);
    float*  losses = (float*) alloc((size_t)T64 * 4);
    int*    done   = (int*)   alloc((size_t)(T64 + 1) * 4);

    const size_t gemm_lds = (size_t)4 * 16 * 32 * 16 * 2;                 // 64 KB
    const size_t lstm_lds = (size_t)(128 * 512 + 512 + 256 + 128 + 128) * 4;

    init_k<<<1, 128, 0, stream>>>(done, losses, (float*)d_out);
    attn_ctx<<<T64, 512, 0, stream>>>(enc, emb, Wattn, battn, words, ctxb);
    gemm_bf16_wmma<<<2048 / 128, 256, gemm_lds, stream>>>(ctxb, Wih, bih, g_in, 2048, H512);
    lstm_seq<<<NWG, 256, lstm_lds, stream>>>(Whh, bhh, h0, c0, g_in, h_all, h2b, done);
    gemm_bf16_wmma<<<(L50K + 127) / 128, 256, gemm_lds, stream>>>(h2b, Wout, bout, logits,
                                                                  L50K, H512);
    loss_k<<<T64, 256, 0, stream>>>(logits, labels, losses);
    sum_k<<<1, T64, 0, stream>>>(losses, (float*)d_out);
}
